// FeatsInitLayer_79542794322611
// MI455X (gfx1250) — matrix operand order repaired and checked by reading, not compiled, not run on hardware
//
#include <hip/hip_runtime.h>
#include <hip/hip_bf16.h>

typedef __attribute__((ext_vector_type(16))) __bf16 v16bf;
typedef __attribute__((ext_vector_type(8)))  __bf16 v8bf;
typedef __attribute__((ext_vector_type(8)))  float  v8f;
typedef __attribute__((ext_vector_type(4)))  float  v4f;

#define WLT_K 208  // padded K stride (elements) for transposed W_lin: 416B, 16B-aligned

__device__ __forceinline__ float swish_f(float v) {
  return v * __builtin_amdgcn_rcpf(1.0f + __expf(-v));
}

__global__ __launch_bounds__(256)
void feats_init_kernel(const int* __restrict__ coords,
                       const float* __restrict__ rbf,
                       const int* __restrict__ node_is,
                       const int* __restrict__ node_js,
                       const float* __restrict__ emb_W,
                       const float* __restrict__ W_rbf0,
                       const float* __restrict__ b_rbf0,
                       const float* __restrict__ W_lin,
                       const float* __restrict__ b_lin,
                       const float* __restrict__ W_rbf1,
                       float* __restrict__ e1_out,
                       float* __restrict__ e2_out,
                       int n_edges) {
  __shared__ __bf16 sWlt[64 * WLT_K];   // W_lin^T  [n][k], bf16
  __shared__ __bf16 sW0t[64 * 16];      // W_rbf0^T [n][k]
  __shared__ __bf16 sW1t[64 * 16];      // W_rbf1^T [n][k]
  __shared__ __bf16 sEmb[95 * 64];      // embedding table, bf16
  __shared__ float  sBl[64];
  __shared__ float  sB0[64];
  __shared__ __bf16 sR0[8][16 * 64];    // per-wave swish(rbf0) staging, row-major

  const int tid = threadIdx.x;

  // ---- cooperative weight staging: f32 global -> bf16 LDS, transposed ----
  for (int idx = tid; idx < 192 * 64; idx += 256) {
    int k = idx >> 6, n = idx & 63;
    sWlt[n * WLT_K + k] = (__bf16)W_lin[idx];
  }
  for (int idx = tid; idx < 16 * 64; idx += 256) {
    int k = idx >> 6, n = idx & 63;
    sW0t[n * 16 + k] = (__bf16)W_rbf0[idx];
    sW1t[n * 16 + k] = (__bf16)W_rbf1[idx];
  }
  for (int idx = tid; idx < 95 * 64; idx += 256)
    sEmb[idx] = (__bf16)emb_W[idx];
  if (tid < 64) { sBl[tid] = b_lin[tid]; sB0[tid] = b_rbf0[tid]; }
  __syncthreads();

  const int wave = tid >> 5;
  const int lane = tid & 31;
  const int lrow = lane & 15;   // A row / C column index within the 16-tile
  const int hi   = lane >> 4;   // selects K half (A/B) and M half (C/D)
  __bf16* const sR0w = &sR0[wave][0];

  const long totalTiles = (long)n_edges >> 4;   // 16 edges per wave-tile
  for (long tile = (long)blockIdx.x * 8 + wave; tile < totalTiles;
       tile += (long)gridDim.x * 8) {
    const long ebase = tile << 4;
    const long erow  = ebase + lrow;

    // species: lanes 0-15 fetch i-side of rows 0-15, lanes 16-31 the j-side
    const int* np = hi ? node_js : node_is;
    const int nid  = np[erow];
    const int spec = coords[nid];
    const int spec_i = __shfl(spec, lrow, 32);
    const int spec_j = __shfl(spec, lrow + 16, 32);

    // ---- A fragment for rbf (16x32 bf16): K=0..15 real, K=16..31 zero pad
    const float* rp = rbf + erow * 16 + hi * 8;
    v4f r0 = *(const v4f*)(rp);
    v4f r1 = *(const v4f*)(rp + 4);
    v16bf a_rbf;
    a_rbf[0] = (__bf16)r0[0]; a_rbf[1] = (__bf16)r0[1];
    a_rbf[2] = (__bf16)r0[2]; a_rbf[3] = (__bf16)r0[3];
    a_rbf[4] = (__bf16)r1[0]; a_rbf[5] = (__bf16)r1[1];
    a_rbf[6] = (__bf16)r1[2]; a_rbf[7] = (__bf16)r1[3];
#pragma unroll
    for (int q = 8; q < 16; ++q) a_rbf[q] = (__bf16)0.0f;

    // ---- rbf0 = swish(rbf@W0 + b0) -> bf16 staging; rbf1p = rbf@W1 kept live
    v8f acc1[4];
#pragma unroll
    for (int t = 0; t < 4; ++t) {
      union { v16bf v; v8bf h[2]; } b0, b1;
      {  // elements 0..7 hold K hi*8..hi*8+7; elements 8..15 (K 16..31) are zero pad
        const __bf16* p0 = sW0t + (t * 16 + lrow) * 16 + hi * 8;
        const __bf16* p1 = sW1t + (t * 16 + lrow) * 16 + hi * 8;
        b0.h[0] = *(const v8bf*)p0;
        b1.h[0] = *(const v8bf*)p1;
#pragma unroll
        for (int q = 8; q < 16; ++q) { b0.v[q] = (__bf16)0.0f; b1.v[q] = (__bf16)0.0f; }
      }
      const float bias0 = sB0[t * 16 + lrow];
      v8f c0, cz;
#pragma unroll
      for (int q = 0; q < 8; ++q) { c0[q] = bias0; cz[q] = 0.0f; }
      v8f racc = __builtin_amdgcn_wmma_f32_16x16x32_bf16(
          false, a_rbf, false, b0.v, (short)0, c0, false, false);
      acc1[t]  = __builtin_amdgcn_wmma_f32_16x16x32_bf16(
          false, a_rbf, false, b1.v, (short)0, cz, false, false);
#pragma unroll
      for (int r = 0; r < 8; ++r) {
        float s = swish_f(racc[r]);
        sR0w[(hi * 8 + r) * 64 + t * 16 + lrow] = (__bf16)s;  // C -> row-major
      }
    }

    // ---- main GEMM: e1 = swish([x_i | x_j | rbf0] @ W_lin + b_lin) ----
    v8f e1acc[4];
#pragma unroll
    for (int t = 0; t < 4; ++t) {
      const float bl = sBl[t * 16 + lrow];
#pragma unroll
      for (int q = 0; q < 8; ++q) e1acc[t][q] = bl;
    }
#pragma unroll
    for (int kcix = 0; kcix < 6; ++kcix) {
      const int kc = kcix * 32;
      const __bf16* asrc;
      int kb;
      if (kcix < 2)      { asrc = sEmb + spec_i * 64; kb = kcix * 32; }
      else if (kcix < 4) { asrc = sEmb + spec_j * 64; kb = (kcix - 2) * 32; }
      else               { asrc = sR0w + lrow * 64;   kb = (kcix - 4) * 32; }
      // A: lane<16 holds K {kc..kc+7, kc+16..23}; lane>=16 holds {+8..15, +24..31}
      union { v16bf v; v8bf h[2]; } af;
      af.h[0] = *(const v8bf*)(asrc + kb + hi * 8);
      af.h[1] = *(const v8bf*)(asrc + kb + 16 + hi * 8);
#pragma unroll
      for (int t = 0; t < 4; ++t) {
        // B: lane = column t*16+lrow; K {kc+hi*8..+7, kc+16+hi*8..+7} per lane
        const __bf16* bp = sWlt + (t * 16 + lrow) * WLT_K + kc + hi * 8;
        union { v16bf v; v8bf h[2]; } bfr;
        bfr.h[0] = *(const v8bf*)bp;
        bfr.h[1] = *(const v8bf*)(bp + 16);
        e1acc[t] = __builtin_amdgcn_wmma_f32_16x16x32_bf16(
            false, af.v, false, bfr.v, (short)0, e1acc[t], false, false);
      }
    }

    // ---- epilogue: swish, e2 = (rbf@W1) * e1, store both ----
#pragma unroll
    for (int t = 0; t < 4; ++t) {
      const int col = t * 16 + lrow;
#pragma unroll
      for (int r = 0; r < 8; ++r) {
        const long row = ebase + hi * 8 + r;
        const float s  = swish_f(e1acc[t][r]);
        const long off = row * 64 + col;
        e1_out[off] = s;
        e2_out[off] = s * acc1[t][r];
      }
    }
  }
}

extern "C" void kernel_launch(void* const* d_in, const int* in_sizes, int n_in,
                              void* d_out, int out_size, void* d_ws, size_t ws_size,
                              hipStream_t stream) {
  const int*   coords  = (const int*)  d_in[0];
  const float* rbf     = (const float*)d_in[1];
  const int*   node_is = (const int*)  d_in[2];
  const int*   node_js = (const int*)  d_in[3];
  const float* emb_W   = (const float*)d_in[4];
  const float* W_rbf0  = (const float*)d_in[5];
  const float* b_rbf0  = (const float*)d_in[6];
  const float* W_lin   = (const float*)d_in[7];
  const float* b_lin   = (const float*)d_in[8];
  const float* W_rbf1  = (const float*)d_in[9];
  const int n_edges = in_sizes[2];              // node_is element count = E

  float* e1 = (float*)d_out;
  float* e2 = e1 + (size_t)n_edges * 64;

  long tiles = (long)n_edges / 16;
  long want  = (tiles + 7) / 8;
  int blocks = (int)(want < 1024 ? (want < 1 ? 1 : want) : 1024);

  feats_init_kernel<<<blocks, 256, 0, stream>>>(
      coords, rbf, node_is, node_js, emb_W, W_rbf0, b_rbf0,
      W_lin, b_lin, W_rbf1, e1, e2, n_edges);
}
